// Model_40097814676122
// MI455X (gfx1250) — matrix operand order repaired and checked by reading, not compiled, not run on hardware
//
#include <hip/hip_runtime.h>
#include <hip/hip_bf16.h>

// ---------------------------------------------------------------------------
// MI455X (gfx1250) implementation.
// All convolutions are lowered to GEMMs on v_wmma_f32_16x16x32_f16 (wave32,
// f16 inputs, f32 accumulate).  Weights are pre-packed into the CDNA5 A-matrix
// fragment layout; activations live in fp16 NCL layout.  Each wave computes a
// 2x16-column B tile against all M tiles (A-fragment reuse, 2*MT WMMAs per
// k-chunk).  Tiny tail ops (attention over 16x64, SE MLPs, classifier) are
// scalar-VALU kernels.
// ---------------------------------------------------------------------------

typedef __attribute__((ext_vector_type(16))) _Float16 v16h;
typedef __attribute__((ext_vector_type(8)))  float    v8f;

#define BN_S 0.9999950000374997f   // (1+1e-5)^-0.5, eval-mode BatchNorm
#define PI_F 3.14159265358979f

__device__ __forceinline__ float selu_f(float x) {
    const float a = 1.6732632423543772f, s = 1.0507009873554805f;
    return x > 0.f ? s * x : s * a * (expf(x) - 1.f);
}
__device__ __forceinline__ float sigm_f(float x) { return 1.f / (1.f + expf(-x)); }
__device__ __forceinline__ float sincn(float z) {    // normalized sinc (jnp.sinc)
    float a = PI_F * z;
    return (fabsf(z) < 1e-8f) ? 1.f : sinf(a) / a;
}

// ---------------------------------------------------------------------------
// Sinc filter bank: one block per filter (70 blocks).
// ---------------------------------------------------------------------------
__global__ __launch_bounds__(256) void k_sinc_filters(const float* __restrict__ low_hz,
                                                      const float* __restrict__ band_hz,
                                                      float* __restrict__ filt) {
    int o = blockIdx.x, t = threadIdx.x;
    __shared__ float red[256];
    float bp = 0.f;
    if (t < 129) {
        float n_   = ((float)t - 64.0f) / 16000.0f;
        float win  = 0.54f - 0.46f * cosf(2.f * PI_F * (float)t / 129.f);
        float low  = fabsf(low_hz[o]) + 1.f;
        float high = fminf(fmaxf(low + fabsf(band_hz[o]), 1.f), 8000.f);
        float fl = low / 16000.f, fh = high / 16000.f;
        bp = (2.f * fh * sincn(2.f * high * n_) - 2.f * fl * sincn(2.f * low * n_)) * win;
    }
    red[t] = fabsf(bp);
    __syncthreads();
    for (int s = 128; s > 0; s >>= 1) { if (t < s) red[t] += red[t + s]; __syncthreads(); }
    if (t < 129) filt[o * 129 + t] = bp / (red[0] + 1e-8f);
}

// ---------------------------------------------------------------------------
// Pack conv weights (O, I, K) f32 -> WMMA A-fragment order, f16.
// Layout: [mtile][kchunk][lane(32)][16 halves]; contiguous K per half-wave
// (lanes 0-15: K 0-15; lanes 16-31: K 16-31), matching the B gather.
// ---------------------------------------------------------------------------
__global__ __launch_bounds__(256) void k_pack(const float* __restrict__ W,
                                              _Float16* __restrict__ out,
                                              int O, int KP, int mtiles, int kchunks) {
    int idx = blockIdx.x * blockDim.x + threadIdx.x;
    int total = mtiles * kchunks * 32 * 16;
    if (idx >= total) return;
    int h    = idx & 15;
    int lane = (idx >> 4) & 31;
    int rest = idx >> 9;
    int kc = rest % kchunks, mt = rest / kchunks;
    int m = mt * 16 + (lane & 15);
    int k = kc * 32 + h + ((lane & 16) ? 16 : 0);
    float v = (m < O && k < KP) ? W[(long)m * KP + k] : 0.f;
    out[idx] = (_Float16)v;
}

// ---------------------------------------------------------------------------
// Generic conv-as-GEMM on WMMA.  out[b,o,n] = act((sum_{i,k} W[o,i,k] *
// in[b,i,n*stride+k-pad] + bias[o]) * BN_S).  MT = #16-row M tiles.
// 8 waves/WG; each wave owns 32 output columns (2 B tiles) and all M tiles.
// ---------------------------------------------------------------------------
struct ConvArgs {
    const _Float16* in;
    const _Float16* wpack;
    const float*    bias;
    _Float16*       out;
    long ibs, obs;            // batch strides (elements)
    int IC, K, Lin, Lout, stride, pad, O, act, kchunks;
};

// Gather one 32x16 B fragment column set (column n, K rows kbase..kbase+15
// for this half-wave).  ic/kk maintained incrementally: no per-element
// integer division; K==1 (1x1 convs) takes a division-free fast path.
__device__ __forceinline__ v16h conv_loadB(const ConvArgs& a, const _Float16* inb,
                                           int n, int kbase, int KP) {
    v16h bf;
    const bool colok = n < a.Lout;
    if (a.K == 1) {
        int pos = n * a.stride - a.pad;
        bool pok = colok && pos >= 0 && pos < a.Lin;
        const _Float16* base = inb + pos;
#pragma unroll
        for (int h = 0; h < 16; ++h) {
            int ic = kbase + h;
            bf[h] = (pok && ic < KP) ? base[(long)ic * a.Lin] : (_Float16)0.f;
        }
    } else {
        int ic = kbase / a.K;
        int kk = kbase - ic * a.K;
#pragma unroll
        for (int h = 0; h < 16; ++h) {
            _Float16 v = (_Float16)0.f;
            if (colok && (kbase + h) < KP) {
                int pos = n * a.stride + kk - a.pad;
                if (pos >= 0 && pos < a.Lin) v = inb[(long)ic * a.Lin + pos];
            }
            bf[h] = v;
            if (++kk == a.K) { kk = 0; ++ic; }
        }
    }
    return bf;
}

template <int MT>
__global__ __launch_bounds__(256) void k_conv_wmma(ConvArgs a) {
    const int b    = blockIdx.y;
    const int wave = threadIdx.x >> 5;
    const int lane = threadIdx.x & 31;
    const int ncol0 = (blockIdx.x * 8 + wave) * 32;
    if (ncol0 >= a.Lout) return;                 // wave-uniform exit: EXEC all-ones at WMMA
    const int n0   = ncol0 + (lane & 15);
    const int n1   = n0 + 16;
    const int koff = (lane & 16) ? 16 : 0;       // contiguous K order (matches k_pack): hi half-wave holds K 16..31
    const int mofs = (lane & 16) ? 8 : 0;        // C/D: hi half-wave holds M rows +8
    const _Float16* inb = a.in + (long)b * a.ibs;
    const int KP = a.IC * a.K;

    v8f acc0[MT] = {};
    v8f acc1[MT] = {};
    for (int kc = 0; kc < a.kchunks; ++kc) {
        const int kbase = kc * 32 + koff;
        v16h bf0 = conv_loadB(a, inb, n0, kbase, KP);
        v16h bf1 = conv_loadB(a, inb, n1, kbase, KP);
#pragma unroll
        for (int mt = 0; mt < MT; ++mt) {
            const v16h af = *(const v16h*)(a.wpack + (((long)mt * a.kchunks + kc) * 32 + lane) * 16);
            acc0[mt] = __builtin_amdgcn_wmma_f32_16x16x32_f16(
                false, af, false, bf0, (short)0, acc0[mt], false, false);
            acc1[mt] = __builtin_amdgcn_wmma_f32_16x16x32_f16(
                false, af, false, bf1, (short)0, acc1[mt], false, false);
        }
    }
    // epilogue: bias + BN + activation, store f16
    _Float16* outb = a.out + (long)b * a.obs;
#pragma unroll
    for (int mt = 0; mt < MT; ++mt) {
#pragma unroll
        for (int e = 0; e < 8; ++e) {
            int m = mt * 16 + e + mofs;
            if (m < a.O) {
                float bsc = a.bias ? a.bias[m] : 0.f;
                if (n0 < a.Lout) {
                    float v = (acc0[mt][e] + bsc) * BN_S;
                    if (a.act == 1)      v = fmaxf(v, 0.f);
                    else if (a.act == 2) v = selu_f(v);
                    outb[(long)m * a.Lout + n0] = (_Float16)v;
                }
                if (n1 < a.Lout) {
                    float v = (acc1[mt][e] + bsc) * BN_S;
                    if (a.act == 1)      v = fmaxf(v, 0.f);
                    else if (a.act == 2) v = selu_f(v);
                    outb[(long)m * a.Lout + n1] = (_Float16)v;
                }
            }
        }
    }
}

// ---------------------------------------------------------------------------
// Elementwise helpers
// ---------------------------------------------------------------------------
__global__ __launch_bounds__(256) void k_f2h(const float* __restrict__ in,
                                             _Float16* __restrict__ out, long n) {
    long i = blockIdx.x * 256L + threadIdx.x;
    if (i < n) out[i] = (_Float16)in[i];
}
__global__ __launch_bounds__(256) void k_copy_h(const _Float16* __restrict__ src, long sbs,
                                                _Float16* __restrict__ dst, long dbs, long nper) {
    long idx = blockIdx.x * 256L + threadIdx.x;
    if (idx >= 16 * nper) return;
    long b = idx / nper, i = idx - b * nper;
    dst[b * dbs + i] = src[b * sbs + i];
}
__global__ __launch_bounds__(256) void k_add_h(const _Float16* __restrict__ x, long xbs,
                                               const _Float16* __restrict__ y, long ybs,
                                               _Float16* __restrict__ o, long obs, long nper) {
    long idx = blockIdx.x * 256L + threadIdx.x;
    if (idx >= 16 * nper) return;
    long b = idx / nper, i = idx - b * nper;
    o[b * obs + i] = (_Float16)((float)x[b * xbs + i] + (float)y[b * ybs + i]);
}

// per-(b,c) mean over L
__global__ __launch_bounds__(256) void k_cmean(const _Float16* __restrict__ x,
                                               float* __restrict__ out, long L) {
    long bc = blockIdx.x;
    const _Float16* p = x + bc * L;
    float s = 0.f;
    for (long i = threadIdx.x; i < L; i += 256) s += (float)p[i];
    __shared__ float red[256];
    red[threadIdx.x] = s; __syncthreads();
    for (int k = 128; k > 0; k >>= 1) { if ((int)threadIdx.x < k) red[threadIdx.x] += red[threadIdx.x + k]; __syncthreads(); }
    if (!threadIdx.x) out[bc] = red[0] / (float)L;
}

// SE MLP: 64 -> 8 (relu) -> 64 (sigmoid), per batch
__global__ __launch_bounds__(64) void k_se(const float* __restrict__ mean,
                                           const float* __restrict__ w1, const float* __restrict__ b1,
                                           const float* __restrict__ w2, const float* __restrict__ b2,
                                           float* __restrict__ se) {
    int b = blockIdx.x, t = threadIdx.x;
    __shared__ float hid[8], m[64];
    m[t] = mean[b * 64 + t]; __syncthreads();
    if (t < 8) {
        float s = b1[t];
        for (int i = 0; i < 64; ++i) s += w1[t * 64 + i] * m[i];
        hid[t] = fmaxf(s, 0.f);
    }
    __syncthreads();
    float s = b2[t];
    for (int i = 0; i < 8; ++i) s += w2[t * 8 + i] * hid[i];
    se[b * 64 + t] = sigm_f(s);
}

// fused relu(conv3*se + residual) + MaxPool1d(3)
__global__ __launch_bounds__(256) void k_pool_se(const _Float16* __restrict__ c3,
                                                 const float* __restrict__ se,
                                                 const _Float16* __restrict__ res,
                                                 _Float16* __restrict__ out, long Lin, long Lp) {
    long idx = blockIdx.x * 256L + threadIdx.x;
    long total = 16L * 64 * Lp;
    if (idx >= total) return;
    long l = idx % Lp, bc = idx / Lp;
    float sc = se[bc];
    const _Float16* cp = c3 + bc * Lin + 3 * l;
    const _Float16* rp = res + bc * Lin + 3 * l;
    float mx = 0.f;
    for (int j = 0; j < 3; ++j)
        mx = fmaxf(mx, fmaxf((float)cp[j] * sc + (float)rp[j], 0.f));
    out[idx] = (_Float16)mx;
}

// AdaptiveAvgPool1d(16) + transpose -> out[b][n][c]
__global__ __launch_bounds__(256) void k_adapt16(const _Float16* __restrict__ x,
                                                 float* __restrict__ out, long L) {
    int idx = blockIdx.x * 256 + threadIdx.x;
    if (idx >= 16 * 16 * 64) return;
    int c = idx & 63, n = (idx >> 6) & 15, b = idx >> 10;
    long seg = L / 16;
    const _Float16* p = x + ((long)b * 64 + c) * L + (long)n * seg;
    float s = 0.f;
    for (long i = 0; i < seg; ++i) s += (float)p[i];
    out[idx] = s / (float)seg;
}

// spec.mean(2), spec.max(2) -> emb columns 128..191, 192..255
__global__ __launch_bounds__(256) void k_spec_stats(const _Float16* __restrict__ x,
                                                    float* __restrict__ dout, long L) {
    int bc = blockIdx.x, b = bc >> 6, c = bc & 63;
    const _Float16* p = x + (long)bc * L;
    float s = 0.f, mx = -1e30f;
    for (long i = threadIdx.x; i < L; i += 256) { float v = (float)p[i]; s += v; mx = fmaxf(mx, v); }
    __shared__ float rs[256], rm[256];
    rs[threadIdx.x] = s; rm[threadIdx.x] = mx; __syncthreads();
    for (int k = 128; k > 0; k >>= 1) {
        if ((int)threadIdx.x < k) {
            rs[threadIdx.x] += rs[threadIdx.x + k];
            rm[threadIdx.x] = fmaxf(rm[threadIdx.x], rm[threadIdx.x + k]);
        }
        __syncthreads();
    }
    if (!threadIdx.x) { dout[b * 256 + 128 + c] = rs[0] / (float)L; dout[b * 256 + 192 + c] = rm[0]; }
}

// ---------------------------------------------------------------------------
// HGAT block (per-batch, everything in LDS; 16x64 tensors, 2 heads)
// ---------------------------------------------------------------------------
struct HgatW {
    const float *sq_w, *sq_b, *sk_w, *sk_b, *sv_w, *sv_b;
    const float *tq_w, *tq_b, *tk_w, *tk_b, *tv_w, *tv_b;
    const float *ca_in_w, *ca_in_b, *ca_out_w, *ca_out_b, *op_w, *op_b;
};

__device__ void lin16(const float* x, int Din, const float* W, const float* bvec,
                      float* y, int Dout) {
    for (int idx = threadIdx.x; idx < 16 * Dout; idx += 256) {
        int nn = idx / Dout, d = idx % Dout;
        float s = bvec[d];
        for (int i = 0; i < Din; ++i) s += x[nn * Din + i] * W[d * Din + i];
        y[idx] = s;
    }
    __syncthreads();
}

__device__ void mha2(const float* Q, const float* Kk, const float* V, float* A, float* O) {
    for (int idx = threadIdx.x; idx < 512; idx += 256) {       // scores, 2 heads
        int hh = idx >> 8, r = (idx >> 4) & 15, c = idx & 15;
        float s = 0.f;
        const float* q = Q + r * 64 + hh * 32;
        const float* k = Kk + c * 64 + hh * 32;
        for (int i = 0; i < 32; ++i) s += q[i] * k[i];
        A[idx] = s * 0.1767766953f;                            // 32^-0.5
    }
    __syncthreads();
    if (threadIdx.x < 32) {                                    // softmax rows
        float* row = A + threadIdx.x * 16;
        float mx = row[0];
        for (int i = 1; i < 16; ++i) mx = fmaxf(mx, row[i]);
        float sm = 0.f;
        for (int i = 0; i < 16; ++i) { row[i] = expf(row[i] - mx); sm += row[i]; }
        for (int i = 0; i < 16; ++i) row[i] /= sm;
    }
    __syncthreads();
    for (int idx = threadIdx.x; idx < 1024; idx += 256) {      // A @ V
        int nn = idx >> 6, d = idx & 63, hh = d >> 5;
        float s = 0.f;
        const float* a = A + hh * 256 + nn * 16;
        for (int k2 = 0; k2 < 16; ++k2) s += a[k2] * V[k2 * 64 + d];
        O[idx] = s;
    }
    __syncthreads();
}

__global__ __launch_bounds__(256) void k_hgat(const float* __restrict__ spec,
                                              const float* __restrict__ temp,
                                              HgatW w, float* __restrict__ gout) {
    int b = blockIdx.x;
    __shared__ float X[1024], XT[1024], S[1024], T[1024], Q[1024], K[1024], V[1024], C[1024];
    __shared__ float SC[2048], A[512];
    for (int i = threadIdx.x; i < 1024; i += 256) { X[i] = spec[b * 1024 + i]; XT[i] = temp[b * 1024 + i]; }
    __syncthreads();
    lin16(X, 64, w.sq_w, w.sq_b, Q, 64);
    lin16(X, 64, w.sk_w, w.sk_b, K, 64);
    lin16(X, 64, w.sv_w, w.sv_b, V, 64);
    mha2(Q, K, V, A, S);
    lin16(XT, 64, w.tq_w, w.tq_b, Q, 64);
    lin16(XT, 64, w.tk_w, w.tk_b, K, 64);
    lin16(XT, 64, w.tv_w, w.tv_b, V, 64);
    mha2(Q, K, V, A, T);
    lin16(S, 64, w.ca_in_w,            w.ca_in_b,       Q, 64);   // wq rows 0-63
    lin16(T, 64, w.ca_in_w + 64 * 64,  w.ca_in_b + 64,  K, 64);   // wk rows 64-127
    lin16(T, 64, w.ca_in_w + 128 * 64, w.ca_in_b + 128, V, 64);   // wv rows 128-191
    mha2(Q, K, V, A, C);
    lin16(C, 64, w.ca_out_w, w.ca_out_b, Q, 64);                  // c projection -> Q
    for (int i = threadIdx.x; i < 1024; i += 256) {               // concat [S, c]
        int nn = i >> 6, d = i & 63;
        SC[nn * 128 + d] = S[i];
        SC[nn * 128 + 64 + d] = Q[i];
    }
    __syncthreads();
    lin16(SC, 128, w.op_w, w.op_b, V, 64);                        // op -> V
    for (int i = threadIdx.x; i < 1024; i += 256) V[i] += X[i];
    __syncthreads();
    if (threadIdx.x < 16) {                                       // layernorm(spec + out)
        float* r = V + threadIdx.x * 64;
        float m = 0.f;
        for (int i = 0; i < 64; ++i) m += r[i];
        m /= 64.f;
        float vv = 0.f;
        for (int i = 0; i < 64; ++i) { float d = r[i] - m; vv += d * d; }
        vv /= 64.f;
        float inv = rsqrtf(vv + 1e-5f);
        float* o = gout + b * 1024 + threadIdx.x * 64;
        for (int i = 0; i < 64; ++i) o[i] = (r[i] - m) * inv;
    }
}

// max graph pool: mean_di max_n ( x[n,di] * sigmoid(x@W^T+b)[n,do] )
__global__ __launch_bounds__(64) void k_mgp(const float* __restrict__ x,
                                            const float* __restrict__ W,
                                            const float* __restrict__ bvec,
                                            float* __restrict__ dout, int off) {
    int b = blockIdx.x, d = threadIdx.x;
    __shared__ float xs[1024];
    for (int i = d; i < 1024; i += 64) xs[i] = x[b * 1024 + i];
    __syncthreads();
    float gate[16];
    for (int nn = 0; nn < 16; ++nn) {
        float s = bvec[d];
        for (int i = 0; i < 64; ++i) s += xs[nn * 64 + i] * W[d * 64 + i];
        gate[nn] = sigm_f(s);
    }
    float acc = 0.f;
    for (int di = 0; di < 64; ++di) {
        float mx = -1e30f;
        for (int nn = 0; nn < 16; ++nn) mx = fmaxf(mx, xs[nn * 64 + di] * gate[nn]);
        acc += mx;
    }
    dout[b * 256 + off + d] = acc / 64.f;
}

__global__ __launch_bounds__(128) void k_classifier(float* __restrict__ dout,
                                                    const float* c1w, const float* c1b,
                                                    const float* c2w, const float* c2b,
                                                    const float* c3w, const float* c3b) {
    int b = blockIdx.x, t = threadIdx.x;
    __shared__ float e[256], z1[128], z2[64];
    for (int i = t; i < 256; i += 128) e[i] = dout[b * 256 + i];
    __syncthreads();
    {
        float s = c1b[t];
        for (int i = 0; i < 256; ++i) s += c1w[t * 256 + i] * e[i];
        z1[t] = selu_f(s * BN_S);
    }
    __syncthreads();
    if (t < 64) {
        float s = c2b[t];
        for (int i = 0; i < 128; ++i) s += c2w[t * 128 + i] * z1[i];
        z2[t] = selu_f(s * BN_S);
    }
    __syncthreads();
    if (t < 2) {
        float s = c3b[t];
        for (int i = 0; i < 64; ++i) s += c3w[t * 64 + i] * z2[i];
        dout[16 * 256 + b * 2 + t] = s;
    }
}

// ---------------------------------------------------------------------------
// Host-side orchestration
// ---------------------------------------------------------------------------
static inline void launch_pack(hipStream_t st, const float* W, _Float16* wp, int O, int KP) {
    int mt = (O + 15) / 16, kc = (KP + 31) / 32;
    int total = mt * kc * 512;
    k_pack<<<(total + 255) / 256, 256, 0, st>>>(W, wp, O, KP, mt, kc);
}

static inline void launch_conv(hipStream_t st, const _Float16* in, long ibs,
                               const _Float16* wp, const float* bias,
                               _Float16* out, long obs,
                               int IC, int K, int Lin, int Lout, int stride, int pad,
                               int O, int act) {
    ConvArgs a;
    a.in = in; a.wpack = wp; a.bias = bias; a.out = out;
    a.ibs = ibs; a.obs = obs;
    a.IC = IC; a.K = K; a.Lin = Lin; a.Lout = Lout;
    a.stride = stride; a.pad = pad; a.O = O; a.act = act;
    a.kchunks = (IC * K + 31) / 32;
    int mt = (O + 15) / 16;
    dim3 g((Lout + 255) / 256, 16);
    if (mt == 5)      k_conv_wmma<5><<<g, 256, 0, st>>>(a);
    else if (mt == 4) k_conv_wmma<4><<<g, 256, 0, st>>>(a);
    else              k_conv_wmma<1><<<g, 256, 0, st>>>(a);
}

extern "C" void kernel_launch(void* const* d_in, const int* in_sizes, int n_in,
                              void* d_out, int out_size, void* d_ws, size_t ws_size,
                              hipStream_t stream) {
    (void)in_sizes; (void)n_in; (void)out_size; (void)ws_size;
    auto F = [&](int i) { return (const float*)d_in[i]; };
    float* out = (float*)d_out;

    const long B = 16, T0 = 62208, L1 = 20736, L2 = 6912, L3 = 2304;
    const long LT1 = 31104, LT2 = 15552, LT3 = 7776;

    char* wsp = (char*)d_ws;
    size_t off = 0;
    auto alloc  = [&](size_t bytes) -> char* { size_t o = (off + 255) & ~(size_t)255; off = o + bytes; return wsp + o; };
    auto allocH = [&](long elems) { return (_Float16*)alloc((size_t)elems * 2); };
    auto allocF = [&](long elems) { return (float*)alloc((size_t)elems * 4); };
    auto packN  = [&](int O, int KP) { return (long)((O + 15) / 16) * ((KP + 31) / 32) * 512; };

    float*    filt    = allocF(70 * 129);
    _Float16* wp_sinc = allocH(packN(70, 129));
    _Float16* wp_r1sc = allocH(packN(64, 70));
    const int cin[3]  = {70, 64, 64};
    _Float16 *wp_c1[3], *wp_s[3][3], *wp_c3[3];
    for (int r = 0; r < 3; ++r) {
        wp_c1[r] = allocH(packN(64, cin[r]));
        for (int j = 0; j < 3; ++j) wp_s[r][j] = allocH(packN(16, 48));
        wp_c3[r] = allocH(packN(64, 64));
    }
    _Float16* wp_t1 = allocH(packN(64, 490));
    _Float16* wp_t2 = allocH(packN(64, 320));
    _Float16* wp_t3 = allocH(packN(64, 192));

    _Float16* xh   = allocH(B * T0);
    _Float16* H    = allocH(B * 70 * T0);
    _Float16* Abuf = allocH(B * 64 * T0);
    _Float16* Bbuf = allocH(B * 64 * T0);
    _Float16* Cbuf = allocH(B * 64 * T0);
    _Float16* Rbuf = allocH(B * 64 * T0);
    _Float16* Sbuf = allocH(B * 16 * T0);
    _Float16* P1   = allocH(B * 64 * L1);
    _Float16* P2   = allocH(B * 64 * L2);
    _Float16* P3   = allocH(B * 64 * L3);
    float* MEANb = allocF(B * 64);
    float* SEb   = allocF(B * 64);
    float* SF = allocF(B * 16 * 64);
    float* TF = allocF(B * 16 * 64);
    float* G1 = allocF(B * 16 * 64);
    float* G2 = allocF(B * 16 * 64);

    // --- weight packing + input conversion -------------------------------
    k_sinc_filters<<<70, 256, 0, stream>>>(F(1), F(2), filt);
    launch_pack(stream, filt, wp_sinc, 70, 129);
    k_f2h<<<(int)((B * T0 + 255) / 256), 256, 0, stream>>>(F(0), xh, B * T0);

    launch_pack(stream, F(3),  wp_c1[0], 64, 70);
    launch_pack(stream, F(4),  wp_s[0][0], 16, 48);
    launch_pack(stream, F(5),  wp_s[0][1], 16, 48);
    launch_pack(stream, F(6),  wp_s[0][2], 16, 48);
    launch_pack(stream, F(7),  wp_c3[0], 64, 64);
    launch_pack(stream, F(12), wp_r1sc, 64, 70);
    launch_pack(stream, F(13), wp_c1[1], 64, 64);
    launch_pack(stream, F(14), wp_s[1][0], 16, 48);
    launch_pack(stream, F(15), wp_s[1][1], 16, 48);
    launch_pack(stream, F(16), wp_s[1][2], 16, 48);
    launch_pack(stream, F(17), wp_c3[1], 64, 64);
    launch_pack(stream, F(22), wp_c1[2], 64, 64);
    launch_pack(stream, F(23), wp_s[2][0], 16, 48);
    launch_pack(stream, F(24), wp_s[2][1], 16, 48);
    launch_pack(stream, F(25), wp_s[2][2], 16, 48);
    launch_pack(stream, F(26), wp_c3[2], 64, 64);
    launch_pack(stream, F(31), wp_t1, 64, 490);
    launch_pack(stream, F(33), wp_t2, 64, 320);
    launch_pack(stream, F(35), wp_t3, 64, 192);

    // --- sinc front-end: h = selu(bn(conv)) -------------------------------
    launch_conv(stream, xh, T0, wp_sinc, nullptr, H, 70 * T0, 1, 129, T0, T0, 1, 64, 70, 2);

    // --- Res2Net block helper --------------------------------------------
    auto res_block = [&](const _Float16* in, long ibs, int ci, long L,
                         _Float16* wpC1, _Float16* wpS0, _Float16* wpS1, _Float16* wpS2,
                         _Float16* wpC3,
                         const float* sew1, const float* seb1,
                         const float* sew2, const float* seb2,
                         const _Float16* resid, _Float16* Pout, long Lp) {
        launch_conv(stream, in, ibs, wpC1, nullptr, Abuf, 64 * L, ci, 1, L, L, 1, 0, 64, 1);
        long tot = B * 16 * L;
        int gb = (int)((tot + 255) / 256);
        k_copy_h<<<gb, 256, 0, stream>>>(Abuf, 64 * L, Bbuf, 64 * L, 16 * L);
        launch_conv(stream, Abuf + 16 * L, 64 * L, wpS0, nullptr, Bbuf + 16 * L, 64 * L, 16, 3, L, L, 1, 1, 16, 1);
        k_add_h<<<gb, 256, 0, stream>>>(Bbuf + 16 * L, 64 * L, Abuf + 32 * L, 64 * L, Sbuf, 16 * L, 16 * L);
        launch_conv(stream, Sbuf, 16 * L, wpS1, nullptr, Bbuf + 32 * L, 64 * L, 16, 3, L, L, 1, 1, 16, 1);
        k_add_h<<<gb, 256, 0, stream>>>(Bbuf + 32 * L, 64 * L, Abuf + 48 * L, 64 * L, Sbuf, 16 * L, 16 * L);
        launch_conv(stream, Sbuf, 16 * L, wpS2, nullptr, Bbuf + 48 * L, 64 * L, 16, 3, L, L, 1, 1, 16, 1);
        launch_conv(stream, Bbuf, 64 * L, wpC3, nullptr, Cbuf, 64 * L, 64, 1, L, L, 1, 0, 64, 0);
        k_cmean<<<(int)(B * 64), 256, 0, stream>>>(Cbuf, MEANb, L);
        k_se<<<(int)B, 64, 0, stream>>>(MEANb, sew1, seb1, sew2, seb2, SEb);
        long tp = B * 64 * Lp;
        k_pool_se<<<(int)((tp + 255) / 256), 256, 0, stream>>>(Cbuf, SEb, resid, Pout, L, Lp);
    };

    // res1 (shortcut conv), res2, res3
    launch_conv(stream, H, 70 * T0, wp_r1sc, nullptr, Rbuf, 64 * T0, 70, 1, T0, T0, 1, 0, 64, 0);
    res_block(H, 70 * T0, 70, T0, wp_c1[0], wp_s[0][0], wp_s[0][1], wp_s[0][2], wp_c3[0],
              F(8), F(9), F(10), F(11), Rbuf, P1, L1);
    res_block(P1, 64 * L1, 64, L1, wp_c1[1], wp_s[1][0], wp_s[1][1], wp_s[1][2], wp_c3[1],
              F(18), F(19), F(20), F(21), P1, P2, L2);
    res_block(P2, 64 * L2, 64, L2, wp_c1[2], wp_s[2][0], wp_s[2][1], wp_s[2][2], wp_c3[2],
              F(27), F(28), F(29), F(30), P2, P3, L3);

    // --- temporal branch (reuses big buffers after spec path) ------------
    launch_conv(stream, H, 70 * T0, wp_t1, F(32), Abuf, 64 * LT1, 70, 7, T0, LT1, 2, 3, 64, 2);
    launch_conv(stream, Abuf, 64 * LT1, wp_t2, F(34), Bbuf, 64 * LT2, 64, 5, LT1, LT2, 2, 2, 64, 2);
    launch_conv(stream, Bbuf, 64 * LT2, wp_t3, F(36), Cbuf, 64 * LT3, 64, 3, LT2, LT3, 2, 1, 64, 2);

    // --- features ---------------------------------------------------------
    k_adapt16<<<64, 256, 0, stream>>>(P3, SF, L3);
    k_adapt16<<<64, 256, 0, stream>>>(Cbuf, TF, LT3);
    k_spec_stats<<<(int)(B * 64), 256, 0, stream>>>(P3, out, L3);

    HgatW w1 = {F(37), F(38), F(39), F(40), F(41), F(42),
                F(43), F(44), F(45), F(46), F(47), F(48),
                F(49), F(50), F(51), F(52), F(53), F(54)};
    k_hgat<<<(int)B, 256, 0, stream>>>(SF, TF, w1, G1);
    HgatW w2 = {F(55), F(56), F(57), F(58), F(59), F(60),
                F(61), F(62), F(63), F(64), F(65), F(66),
                F(67), F(68), F(69), F(70), F(71), F(72)};
    k_hgat<<<(int)B, 256, 0, stream>>>(G1, TF, w2, G2);

    k_mgp<<<(int)B, 64, 0, stream>>>(G2, F(73), F(74), out, 0);
    k_mgp<<<(int)B, 64, 0, stream>>>(TF, F(75), F(76), out, 64);
    k_classifier<<<(int)B, 128, 0, stream>>>(out, F(77), F(78), F(79), F(80), F(81), F(82));
}
